// NetBA_56315611185914
// MI455X (gfx1250) — matrix operand order repaired and checked by reading, not compiled, not run on hardware
//
#include <hip/hip_runtime.h>

typedef __bf16 bf16;
typedef bf16  v16bf __attribute__((ext_vector_type(16)));
typedef float v8f   __attribute__((ext_vector_type(8)));

#define NN   100000
#define NE   800000
#define NG   128
#define INC  66
#define K1P  96
#define DIM  256
#define OUTC 128
#define BNINV 0.99999500003750f   // 1/sqrt(1+1e-5)

union ABReg { v16bf v; uint4 q[2]; };

__device__ __forceinline__ void pk_add_bf16(bf16* p, unsigned int v) {
  asm volatile("global_atomic_pk_add_bf16 %0, %1, off" :: "v"(p), "v"(v) : "memory");
}
__device__ __forceinline__ void atom_add_f32(float* p, float v) {
  asm volatile("global_atomic_add_f32 %0, %1, off" :: "v"(p), "v"(v) : "memory");
}

// ---------------- weight packing: f32 W[K][N] -> bf16 WMMA-B tiles ----------------
// tile = nt*KT + kt ; per tile: 32 lanes x 16 bf16 (lane: n = nt*16 + (l&15),
// k = kt*32 + (l<16 ? 0 : 16) + e, e=0..15)
__global__ __launch_bounds__(256) void pack_w(const float* __restrict__ W,
                                              bf16* __restrict__ dst,
                                              int K, int Kpad, int N) {
  int t = blockIdx.x * 256 + threadIdx.x;
  int KT = Kpad >> 5, NT = N >> 4;
  if (t >= KT * NT * 32) return;
  int lane = t & 31;
  int tile = t >> 5;
  int kt = tile % KT, nt = tile / KT;
  int n = nt * 16 + (lane & 15);
  int kbase = kt * 32 + ((lane < 16) ? 0 : 16);
  bf16* o = dst + (size_t)tile * 512 + lane * 16;
#pragma unroll
  for (int e = 0; e < 16; ++e) {
    int k = kbase + e;
    float v = (k < K) ? W[(size_t)k * N + n] : 0.0f;
    o[e] = (bf16)v;
  }
}

// ---------------- convert/pad x to bf16, duplicate into agg0 ----------------
__global__ __launch_bounds__(256) void cvt_x(const float* __restrict__ x,
                                             bf16* __restrict__ xb,
                                             bf16* __restrict__ agg0) {
  int i = blockIdx.x * 256 + threadIdx.x;
  if (i >= NN * K1P) return;
  int node = i / K1P, c = i - node * K1P;
  float v = (c < INC) ? x[(size_t)node * INC + c] : 0.0f;
  bf16 b = (bf16)v;
  xb[i] = b;
  agg0[i] = b;
}

// ---------------- edge scatter-add: one wave per edge ----------------
__global__ __launch_bounds__(256) void edge_agg(const long long* __restrict__ ei,
                                                const bf16* __restrict__ h,
                                                bf16* __restrict__ agg, int width) {
  int e = blockIdx.x * 8 + (threadIdx.x >> 5);
  if (e >= NE) return;
  int lane = threadIdx.x & 31;
  long long src = ei[e];
  long long dst = ei[NE + e];
  int c = lane * 8;
  if (c < width) {
    uint4 v = *(const uint4*)(h + (size_t)src * width + c);
    bf16* p = agg + (size_t)dst * width + c;
    pk_add_bf16(p + 0, v.x);
    pk_add_bf16(p + 2, v.y);
    pk_add_bf16(p + 4, v.z);
    pk_add_bf16(p + 6, v.w);
  }
}

// ---------------- WMMA GEMM + fused bias/BN/ReLU epilogue ----------------
// A: [M, K] bf16 (row-major), Wp: packed B tiles, grid.x = M/16, block = 256.
// mode 0: bf16 -> o0 ; mode 1: bf16 -> o0 and o1 ; mode 2: f32 -> of
__global__ __launch_bounds__(256) void gemm_wmma(
    const bf16* __restrict__ A, int K, int KT,
    const bf16* __restrict__ Wp,
    const float* __restrict__ bias,
    const float* __restrict__ gamma,
    const float* __restrict__ beta,
    int N, int relu, int mode,
    bf16* __restrict__ o0, bf16* __restrict__ o1, float* __restrict__ of) {
  __shared__ __align__(16) bf16 As[16 * 264];
  const int astride = K + 8;                 // 104 or 264 bf16 -> conflict-friendly
  const int tid  = threadIdx.x;
  const int lane = tid & 31;
  const int wid  = tid >> 5;
  const size_t m0 = (size_t)blockIdx.x * 16;

  // stage A tile (16 x K) into LDS
  const int cpr = K >> 3;                    // uint4 chunks per row
  const int chunks = cpr << 4;
  for (int i = tid; i < chunks; i += 256) {
    int row = i / cpr;
    int c8  = i - row * cpr;
    uint4 v = *(const uint4*)(A + (m0 + row) * (size_t)K + c8 * 8);
    *(uint4*)&As[row * astride + c8 * 8] = v;
  }
  __syncthreads();

  const int mrow = lane & 15;
  const int hi   = lane >> 4;
  const int aoff = hi * 16;                  // A layout: K {0..15|16..31} contiguous per lane half
  const int mloc = hi << 3;                  // C/D layout: lanes 16-31 hold M+8
  const int NT = N >> 4;

  for (int nt = wid; nt < NT; nt += 8) {
    v8f acc = {};
    const bf16* bp = Wp + ((size_t)nt * KT) * 512 + lane * 16;
    for (int kt = 0; kt < KT; ++kt) {
      ABReg a, b;
      const bf16* ap = &As[mrow * astride + kt * 32 + aoff];
      a.q[0] = *(const uint4*)(ap);
      a.q[1] = *(const uint4*)(ap + 8);
      b.q[0] = *(const uint4*)(bp);
      b.q[1] = *(const uint4*)(bp + 8);
      bp += 512;
      acc = __builtin_amdgcn_wmma_f32_16x16x32_bf16(false, a.v, false, b.v,
                                                    (short)0, acc, false, false);
    }
    int n = (nt << 4) + mrow;
    float bs = bias[n];
    float sc = 1.0f, sh = 0.0f;
    if (gamma) { sc = gamma[n] * BNINV; sh = beta[n]; }
#pragma unroll
    for (int r = 0; r < 8; ++r) {
      float v = (acc[r] + bs) * sc + sh;
      if (relu) v = v > 0.0f ? v : 0.0f;
      size_t idx = (m0 + mloc + r) * (size_t)N + n;
      if (mode == 2) {
        of[idx] = v;
      } else {
        bf16 bv = (bf16)v;
        o0[idx] = bv;
        if (mode == 1) o1[idx] = bv;
      }
    }
  }
}

// ---------------- pooling ----------------
__global__ __launch_bounds__(256) void zero_f32(float* __restrict__ p, int n) {
  int i = blockIdx.x * 256 + threadIdx.x;
  if (i < n) p[i] = 0.0f;
}
__global__ __launch_bounds__(256) void pool_sum(const bf16* __restrict__ h,
                                                const long long* __restrict__ batch,
                                                float* __restrict__ sums,
                                                float* __restrict__ cnt) {
  int i = blockIdx.x * 256 + threadIdx.x;
  int node = i >> 8, c = i & 255;
  if (node >= NN) return;
  int g = (int)batch[node];
  atom_add_f32(&sums[g * 256 + c], (float)h[i]);
  if (c == 0) atom_add_f32(&cnt[g], 1.0f);
}
__global__ __launch_bounds__(256) void pool_fin(const float* __restrict__ sums,
                                                const float* __restrict__ cnt,
                                                bf16* __restrict__ pb) {
  int i = blockIdx.x * 256 + threadIdx.x;
  if (i >= NG * 256) return;
  int g = i >> 8;
  float d = cnt[g];
  d = d > 1.0f ? d : 1.0f;
  pb[i] = (bf16)(sums[i] / d);
}

// ---------------- driver ----------------
extern "C" void kernel_launch(void* const* d_in, const int* in_sizes, int n_in,
                              void* d_out, int out_size, void* d_ws, size_t ws_size,
                              hipStream_t stream) {
  (void)in_sizes; (void)n_in; (void)out_size; (void)ws_size;
  const float*     x     = (const float*)d_in[0];
  const long long* ei    = (const long long*)d_in[1];
  const long long* batch = (const long long*)d_in[2];
  const float *W1 = (const float*)d_in[3],  *b1  = (const float*)d_in[4];
  const float *g1 = (const float*)d_in[5],  *be1 = (const float*)d_in[6];
  const float *W2 = (const float*)d_in[7],  *b2  = (const float*)d_in[8];
  const float *g2 = (const float*)d_in[9],  *be2 = (const float*)d_in[10];
  const float *W3 = (const float*)d_in[11], *b3  = (const float*)d_in[12];
  const float *g3 = (const float*)d_in[13], *be3 = (const float*)d_in[14];
  const float *Wl1 = (const float*)d_in[15], *bl1 = (const float*)d_in[16];
  const float *Wl2 = (const float*)d_in[17], *bl2 = (const float*)d_in[18];

  char* ws = (char*)d_ws;
  size_t off = 0;
  auto alloc = [&](size_t bytes) -> void* {
    void* p = ws + off;
    off = (off + bytes + 255) & ~(size_t)255;
    return p;
  };
  bf16* XB  = (bf16*)alloc((size_t)NN * K1P * 2);
  bf16* AG0 = (bf16*)alloc((size_t)NN * K1P * 2);
  bf16* BA  = (bf16*)alloc((size_t)NN * DIM * 2);  // h1, later h3
  bf16* BB  = (bf16*)alloc((size_t)NN * DIM * 2);  // agg1, later h4
  bf16* BC  = (bf16*)alloc((size_t)NN * DIM * 2);  // h2
  bf16* BD  = (bf16*)alloc((size_t)NN * DIM * 2);  // agg2
  bf16* WP1  = (bf16*)alloc((size_t)3 * 16 * 512 * 2);
  bf16* WP2  = (bf16*)alloc((size_t)8 * 16 * 512 * 2);
  bf16* WP3  = (bf16*)alloc((size_t)8 * 16 * 512 * 2);
  bf16* WPL1 = (bf16*)alloc((size_t)8 * 16 * 512 * 2);
  bf16* WPL2 = (bf16*)alloc((size_t)8 * 8 * 512 * 2);
  float* PSUM = (float*)alloc((size_t)(NG * 256 + NG) * 4);
  float* CNT  = PSUM + NG * 256;
  bf16* PB = (bf16*)alloc((size_t)NG * 256 * 2);
  bf16* H5 = (bf16*)alloc((size_t)NG * 256 * 2);

  // pack weights (bf16 WMMA-B tiles)
  pack_w<<<6, 256, 0, stream>>>(W1,  WP1,  INC, K1P, DIM);
  pack_w<<<16, 256, 0, stream>>>(W2,  WP2,  DIM, DIM, DIM);
  pack_w<<<16, 256, 0, stream>>>(W3,  WP3,  DIM, DIM, DIM);
  pack_w<<<16, 256, 0, stream>>>(Wl1, WPL1, DIM, DIM, DIM);
  pack_w<<<8, 256, 0, stream>>>(Wl2, WPL2, DIM, DIM, OUTC);

  // layer 1
  cvt_x<<<(NN * K1P + 255) / 256, 256, 0, stream>>>(x, XB, AG0);
  edge_agg<<<NE / 8, 256, 0, stream>>>(ei, XB, AG0, K1P);
  gemm_wmma<<<NN / 16, 256, 0, stream>>>(AG0, K1P, K1P / 32, WP1, b1, g1, be1,
                                         DIM, 1, 1, BA, BB, nullptr);
  // layer 2
  edge_agg<<<NE / 8, 256, 0, stream>>>(ei, BA, BB, DIM);
  gemm_wmma<<<NN / 16, 256, 0, stream>>>(BB, DIM, DIM / 32, WP2, b2, g2, be2,
                                         DIM, 1, 1, BC, BD, nullptr);
  // layer 3 (no ReLU)
  edge_agg<<<NE / 8, 256, 0, stream>>>(ei, BC, BD, DIM);
  gemm_wmma<<<NN / 16, 256, 0, stream>>>(BD, DIM, DIM / 32, WP3, b3, g3, be3,
                                         DIM, 0, 0, BA, nullptr, nullptr);
  // node-level Linear + ReLU
  gemm_wmma<<<NN / 16, 256, 0, stream>>>(BA, DIM, DIM / 32, WPL1, bl1, nullptr, nullptr,
                                         DIM, 1, 0, BB, nullptr, nullptr);
  // global mean pool
  zero_f32<<<(NG * 256 + NG + 255) / 256, 256, 0, stream>>>(PSUM, NG * 256 + NG);
  pool_sum<<<NN, 256, 0, stream>>>(BB, batch, PSUM, CNT);
  pool_fin<<<NG, 256, 0, stream>>>(PSUM, CNT, PB);
  // head
  gemm_wmma<<<NG / 16, 256, 0, stream>>>(PB, DIM, DIM / 32, WPL1, bl1, nullptr, nullptr,
                                         DIM, 1, 0, H5, nullptr, nullptr);
  gemm_wmma<<<NG / 16, 256, 0, stream>>>(H5, DIM, DIM / 32, WPL2, bl2, nullptr, nullptr,
                                         OUTC, 0, 2, nullptr, nullptr, (float*)d_out);
}
